// RQVAE_43121471652243
// MI455X (gfx1250) — matrix operand order repaired and checked by reading, not compiled, not run on hardware
//
#include <hip/hip_runtime.h>

// ---------------------------------------------------------------------------
// Types
// ---------------------------------------------------------------------------
typedef __attribute__((ext_vector_type(16))) __bf16         v16bf;
typedef __attribute__((ext_vector_type(8)))  __bf16         v8bf;
typedef __attribute__((ext_vector_type(8)))  float          v8f;
typedef __attribute__((ext_vector_type(4)))  float          f32x4;
typedef __attribute__((ext_vector_type(4)))  unsigned int   u32x4;
typedef __attribute__((ext_vector_type(4)))  unsigned short u16x4;
typedef __attribute__((ext_vector_type(4)))  int            i32x4;
typedef __attribute__((ext_vector_type(8)))  int            i32x8;

#define B_ROWS   65536
#define IN_DIM_  768
#define E_DIM_   128
#define NUM_EMB_ 256
#define CODE_LEN_ 4
// rq_loss scale: (1+beta) / (CODE_LEN * B * E_DIM)
#define LOSS_SCALE (1.25f / (4.0f * 65536.0f * 128.0f))

// Tensor Data Mover path (6-arg builtin on this toolchain: clang-23 lane)
#if defined(__has_builtin)
#if __has_builtin(__builtin_amdgcn_tensor_load_to_lds) && \
    __has_builtin(__builtin_amdgcn_s_wait_tensorcnt)
#define USE_TDM 1
#endif
#endif

__device__ __forceinline__ unsigned short f2bf(float f) {
    unsigned u = __builtin_bit_cast(unsigned, f);
    unsigned r = u + 0x7FFFu + ((u >> 16) & 1u);   // round-to-nearest-even
    return (unsigned short)(r >> 16);
}

#ifdef USE_TDM
// byte offset of a static __shared__ object within the workgroup LDS
__device__ __forceinline__ unsigned lds_off(void* p) {
    return (unsigned)(unsigned long long)(__attribute__((address_space(3))) void*)p;
}
#endif

// ---------------------------------------------------------------------------
// f32 -> bf16 conversion (vector x4) for the activation input x
// ---------------------------------------------------------------------------
__global__ void cvt_f32_to_bf16_v4(const f32x4* __restrict__ in,
                                   u16x4* __restrict__ out, long n4) {
    long i = (long)blockIdx.x * blockDim.x + threadIdx.x;
    if (i < n4) {
        f32x4 v = in[i];
        u16x4 o;
        o[0] = f2bf(v[0]); o[1] = f2bf(v[1]);
        o[2] = f2bf(v[2]); o[3] = f2bf(v[3]);
        out[i] = o;
    }
}

// ---------------------------------------------------------------------------
// Weight convert + transpose: W[K][N] f32 -> Wt[N][K] bf16 (32x32 LDS tiles)
// ---------------------------------------------------------------------------
__global__ __launch_bounds__(256)
void cvt_transpose_kernel(const float* __restrict__ W,
                          unsigned short* __restrict__ Wt, int K, int N) {
    __shared__ float tile[32][33];
    const int bn = blockIdx.x * 32;          // N tile
    const int bk = blockIdx.y * 32;          // K tile
    const int tx = threadIdx.x & 31;
    const int ty = threadIdx.x >> 5;         // 0..7
#pragma unroll
    for (int j = 0; j < 32; j += 8)
        tile[ty + j][tx] = W[(size_t)(bk + ty + j) * N + bn + tx];
    __syncthreads();
#pragma unroll
    for (int j = 0; j < 32; j += 8)
        Wt[(size_t)(bn + ty + j) * K + bk + tx] = f2bf(tile[tx][ty + j]);
}

// ---------------------------------------------------------------------------
// codebooks [4][256][128] -> cbT [4][128][256]  (tiny)
// ---------------------------------------------------------------------------
__global__ void transpose_cb_kernel(const float* __restrict__ cb,
                                    float* __restrict__ cbT) {
    int i = blockIdx.x * 256 + threadIdx.x;      // indexes output
    int c  = i & 255;
    int ld = i >> 8;
    int d  = ld & 127;
    int l  = ld >> 7;
    cbT[i] = cb[((size_t)(l * 256 + c)) * 128 + d];
}

__global__ void zero_loss_kernel(float* p) {
    if (threadIdx.x == 0 && blockIdx.x == 0) *p = 0.0f;
}

// ---------------------------------------------------------------------------
// Tiled bf16 GEMM:  C[M,N] = A[M,K] * Wt[N,K]^T + bias (+ReLU)
//   A: bf16 row-major [M][K];  Wt: bf16 row-major [N][K] (pre-transposed)
//   Tile: BM=128 BN=128 BK=64, 256 threads = 8 waves (2 along M x 4 along N),
//   each wave owns 64x32 -> 4x2 WMMA f32 accumulators, 16 WMMA per K-tile.
//   Staging: TDM tensor_load_to_lds (2D tile 64x128, LDS rows padded 64->72)
// ---------------------------------------------------------------------------
#define BM 128
#define BN 128
#define BK 64
#define LDT 72   // padded LDS row stride (elements) -> 144B, 16B-aligned frags

template <bool RELU, bool WBF, bool WF32>
__global__ __launch_bounds__(256)
void gemm_bf16_kernel(const unsigned short* __restrict__ A,
                      const unsigned short* __restrict__ Wt,
                      const float* __restrict__ bias,
                      unsigned short* __restrict__ Cb,
                      float* __restrict__ Cf,
                      int M, int N, int K) {
    __shared__ unsigned short As[2][BM * LDT];   // [m][k]
    __shared__ unsigned short Bs[2][BN * LDT];   // [n][k]

    const int tid  = threadIdx.x;
    const int lane = tid & 31;
    const int wave = tid >> 5;
    const int wm   = wave & 1;    // wave row (M)
    const int wn   = wave >> 1;   // wave col (N)
    const int bm   = blockIdx.y * BM;
    const int bn   = blockIdx.x * BN;
    const int nk   = K / BK;

    v8f acc[4][2];
#pragma unroll
    for (int mi = 0; mi < 4; ++mi)
#pragma unroll
        for (int ni = 0; ni < 2; ++ni)
#pragma unroll
            for (int r = 0; r < 8; ++r) acc[mi][ni][r] = 0.0f;

#ifdef USE_TDM
    // ---- Tensor Data Mover staging ----
    // D# group1 (shared by A/B tiles): data_size=2B, pad every 32 DWORDs by
    // 4 DWORDs (row 64 elem -> LDS stride 72 elem), tensor 64 x 128,
    // tile 64 x 128, dim0 stride = K elements.
    const unsigned g1w0 = (1u << 16) | (1u << 20) | (4u << 22) | (3u << 25);
    i32x8 g1;
    g1[0] = (int)g1w0;
    g1[1] = (int)(64u << 16);          // tensor_dim0[15:0] in [63:48]
    g1[2] = (int)(128u << 16);         // tensor_dim0 hi | tensor_dim1 lo
    g1[3] = (int)(64u << 16);          // tensor_dim1 hi | tile_dim0
    g1[4] = 128;                       // tile_dim1 (tile_dim2 = 0)
    g1[5] = K;                         // tensor_dim0_stride[31:0]
    g1[6] = 0;
    g1[7] = 0;
    const i32x4 gz4 = {0, 0, 0, 0};    // groups 2/3: tile_dim3/4 = 0 (unused)
    const i32x8 gz8 = {0, 0, 0, 0, 0, 0, 0, 0};  // extended words (unused)

    unsigned ldsA0 = lds_off(&As[0][0]), ldsA1 = lds_off(&As[1][0]);
    unsigned ldsB0 = lds_off(&Bs[0][0]), ldsB1 = lds_off(&Bs[1][0]);

    auto tdm_issue = [&](int kt, int buf) {
        unsigned long long ga =
            (unsigned long long)(A + (size_t)bm * K + kt * BK);
        unsigned long long gb =
            (unsigned long long)(Wt + (size_t)bn * K + kt * BK);
        u32x4 g0a, g0b;
        g0a[0] = 1u;  // count=1
        g0a[1] = buf ? ldsA1 : ldsA0;
        g0a[2] = (unsigned)(ga & 0xFFFFFFFFu);
        g0a[3] = (unsigned)((ga >> 32) & 0x01FFFFFFu) | 0x80000000u;  // type=2
        g0b[0] = 1u;
        g0b[1] = buf ? ldsB1 : ldsB0;
        g0b[2] = (unsigned)(gb & 0xFFFFFFFFu);
        g0b[3] = (unsigned)((gb >> 32) & 0x01FFFFFFu) | 0x80000000u;
        __builtin_amdgcn_tensor_load_to_lds(g0a, g1, gz4, gz4, gz8, 0);
        __builtin_amdgcn_tensor_load_to_lds(g0b, g1, gz4, gz4, gz8, 0);
    };

    if (wave == 0) tdm_issue(0, 0);
    __builtin_amdgcn_s_wait_tensorcnt(0);
    __syncthreads();
#else
    // ---- fallback: per-thread copy with register prefetch ----
    u32x4 ra[4], rb[4];
    auto loadT = [&](const unsigned short* base, int kt, u32x4* r) {
        int k0 = kt * BK;
#pragma unroll
        for (int j = 0; j < 4; ++j) {
            int ch  = tid + 256 * j;
            int row = ch >> 3, c8 = ch & 7;
            r[j] = *(const u32x4*)(base + (size_t)row * K + k0 + c8 * 8);
        }
    };
    auto storeT = [&](unsigned short* lds, const u32x4* r) {
#pragma unroll
        for (int j = 0; j < 4; ++j) {
            int ch  = tid + 256 * j;
            int row = ch >> 3, c8 = ch & 7;
            *(u32x4*)(&lds[row * LDT + c8 * 8]) = r[j];
        }
    };
    const unsigned short* Abase = A + (size_t)bm * K;
    const unsigned short* Bbase = Wt + (size_t)bn * K;
    loadT(Abase, 0, ra); loadT(Bbase, 0, rb);
    storeT(As[0], ra);   storeT(Bs[0], rb);
    __syncthreads();
#endif

    const int arow = lane & 15;
    const int alo  = (lane < 16) ? 0 : 8;   // K sub-chunk base (A layout)
    const int bhal = (lane < 16) ? 0 : 8;   // K sub-chunk base (B layout, same as A)

    for (int kt = 0; kt < nk; ++kt) {
        const int cur = kt & 1, nxt = cur ^ 1;
#ifdef USE_TDM
        if (kt + 1 < nk && wave == 0) tdm_issue(kt + 1, nxt);
#else
        if (kt + 1 < nk) { loadT(Abase, kt + 1, ra); loadT(Bbase, kt + 1, rb); }
#endif
        const unsigned short* as = As[cur];
        const unsigned short* bs = Bs[cur];

#pragma unroll
        for (int ks = 0; ks < 2; ++ks) {       // two 32-deep WMMA sub-steps
            const int ko = ks * 32;
            v16bf af[4], bf[2];
#pragma unroll
            for (int mi = 0; mi < 4; ++mi) {
                int r0 = (wm * 64 + mi * 16 + arow) * LDT + ko;
                v8bf lo = *(const v8bf*)(const void*)(as + r0 + alo);
                v8bf hi = *(const v8bf*)(const void*)(as + r0 + alo + 16);
                af[mi] = __builtin_shufflevector(lo, hi, 0,1,2,3,4,5,6,7,8,9,10,11,12,13,14,15);
            }
#pragma unroll
            for (int ni = 0; ni < 2; ++ni) {
                int r0 = (wn * 32 + ni * 16 + arow) * LDT + ko;
                v8bf lo = *(const v8bf*)(const void*)(bs + r0 + bhal);
                v8bf hi = *(const v8bf*)(const void*)(bs + r0 + bhal + 16);
                bf[ni] = __builtin_shufflevector(lo, hi, 0,1,2,3,4,5,6,7,8,9,10,11,12,13,14,15);
            }
#pragma unroll
            for (int mi = 0; mi < 4; ++mi)
#pragma unroll
                for (int ni = 0; ni < 2; ++ni)
                    acc[mi][ni] = __builtin_amdgcn_wmma_f32_16x16x32_bf16(
                        false, af[mi], false, bf[ni], (short)0, acc[mi][ni],
                        false, false);
        }

        if (kt + 1 < nk) {
#ifdef USE_TDM
            __builtin_amdgcn_s_wait_tensorcnt(0);
#else
            storeT(As[nxt], ra); storeT(Bs[nxt], rb);
#endif
            __syncthreads();
        }
    }

    // ---- epilogue: bias (+ReLU), branch-free via template flags ----
#pragma unroll
    for (int ni = 0; ni < 2; ++ni) {
        const int col = bn + wn * 32 + ni * 16 + (lane & 15);
        const float bv = bias[col];
#pragma unroll
        for (int mi = 0; mi < 4; ++mi) {
            const int row0 = bm + wm * 64 + mi * 16 + ((lane >> 4) << 3);
#pragma unroll
            for (int r = 0; r < 8; ++r) {
                float v = acc[mi][ni][r] + bv;
                if (RELU) v = fmaxf(v, 0.0f);
                const size_t off = (size_t)(row0 + r) * N + col;
                if (WF32) Cf[off] = v;
                if (WBF)  Cb[off] = f2bf(v);
            }
        }
    }
}

// ---------------------------------------------------------------------------
// Residual quantization: per block 64 rows.
//   res in LDS (f32), codebook level staged in LDS in 2 chunks of 128 codes.
//   score = ||c||^2 - 2 r.c ; argmin with lowest-index tie-break.
//   loss += (1+beta)*sum((q-r)^2)/(L*B*D) via atomics; xq = z - res_final.
// ---------------------------------------------------------------------------
#define RQ_ROWS 64

__global__ __launch_bounds__(256)
void rq_kernel(const float* __restrict__ z,
               const float* __restrict__ cbT,   // [4][128][256]
               const float* __restrict__ cb,    // [4][256][128]
               unsigned short* __restrict__ xq, // [B][128] bf16
               int* __restrict__ indices,       // [B][4]
               float* __restrict__ loss) {
    __shared__ float res[RQ_ROWS * 128];   // 32KB
    __shared__ float cbl[128 * 128];       // 64KB (one chunk of 128 codes)
    __shared__ float cnorm[128];
    __shared__ float bestS[RQ_ROWS];
    __shared__ int   bestI[RQ_ROWS];

    const int tid  = threadIdx.x;
    const int lane = tid & 31;
    const int wave = tid >> 5;
    const size_t row0 = (size_t)blockIdx.x * RQ_ROWS;

    for (int i = tid; i < RQ_ROWS * 128; i += 256)
        res[i] = z[row0 * 128 + i];

    for (int l = 0; l < CODE_LEN_; ++l) {
        if (tid < RQ_ROWS) bestS[tid] = 3.0e38f;

        for (int cc = 0; cc < 2; ++cc) {
            __syncthreads();
            for (int i = tid; i < 128 * 32; i += 256) {   // float4 units
                int d = i >> 5, c4 = i & 31;
                ((f32x4*)cbl)[i] =
                    *(const f32x4*)(cbT + ((size_t)(l * 128 + d)) * 256 +
                                    cc * 128 + c4 * 4);
            }
            __syncthreads();
            if (tid < 128) {
                float s = 0.0f;
                for (int d = 0; d < 128; ++d) {
                    float v = cbl[d * 128 + tid];
                    s += v * v;
                }
                cnorm[tid] = s;
            }
            __syncthreads();

            for (int rg = 0; rg < 2; ++rg) {
                int rbase = wave * 8 + rg * 4;
                float acc[4][4];
#pragma unroll
                for (int i = 0; i < 4; ++i)
#pragma unroll
                    for (int rr = 0; rr < 4; ++rr) acc[i][rr] = 0.0f;

                for (int d = 0; d < 128; ++d) {
                    float cv0 = cbl[d * 128 + lane];
                    float cv1 = cbl[d * 128 + 32 + lane];
                    float cv2 = cbl[d * 128 + 64 + lane];
                    float cv3 = cbl[d * 128 + 96 + lane];
#pragma unroll
                    for (int rr = 0; rr < 4; ++rr) {
                        float rv = res[(rbase + rr) * 128 + d];
                        acc[0][rr] = fmaf(rv, cv0, acc[0][rr]);
                        acc[1][rr] = fmaf(rv, cv1, acc[1][rr]);
                        acc[2][rr] = fmaf(rv, cv2, acc[2][rr]);
                        acc[3][rr] = fmaf(rv, cv3, acc[3][rr]);
                    }
                }
#pragma unroll
                for (int rr = 0; rr < 4; ++rr) {
                    float bs = 3.0e38f; int bi = 0x7FFFFFFF;
#pragma unroll
                    for (int i = 0; i < 4; ++i) {
                        int c = cc * 128 + i * 32 + lane;
                        float s = cnorm[i * 32 + lane] - 2.0f * acc[i][rr];
                        if (s < bs || (s == bs && c < bi)) { bs = s; bi = c; }
                    }
                    for (int off = 16; off > 0; off >>= 1) {
                        float os = __shfl_xor(bs, off, 32);
                        int   oi = __shfl_xor(bi, off, 32);
                        if (os < bs || (os == bs && oi < bi)) { bs = os; bi = oi; }
                    }
                    if (lane == 0) {
                        int lr = rbase + rr;
                        if (bs < bestS[lr]) { bestS[lr] = bs; bestI[lr] = bi; }
                        else if (bs == bestS[lr] && bi < bestI[lr]) bestI[lr] = bi;
                    }
                }
            }
        }

        float errw = 0.0f;
        for (int r = wave * 8; r < wave * 8 + 8; ++r) {
            int idx = bestI[r];
            if (lane == 0) indices[(row0 + r) * 4 + l] = idx;
            const float* qrow = cb + ((size_t)l * NUM_EMB_ + idx) * 128;
#pragma unroll
            for (int j = 0; j < 4; ++j) {
                int d = lane + 32 * j;
                float qv = qrow[d];
                float rv = res[r * 128 + d];
                float df = qv - rv;
                errw = fmaf(df, df, errw);
                res[r * 128 + d] = rv - qv;
            }
        }
        for (int off = 16; off > 0; off >>= 1) errw += __shfl_xor(errw, off, 32);
        if (lane == 0) atomicAdd(loss, errw * LOSS_SCALE);
        __syncthreads();
    }

    for (int i = tid; i < RQ_ROWS * 128; i += 256) {
        float zx = z[row0 * 128 + i];
        xq[row0 * 128 + i] = f2bf(zx - res[i]);
    }
}

// ---------------------------------------------------------------------------
// Host side
// ---------------------------------------------------------------------------
static void run_cvt(const float* src, unsigned short* dst, size_t n,
                    hipStream_t s) {
    size_t n4 = n / 4;
    dim3 g((unsigned)((n4 + 255) / 256));
    cvt_f32_to_bf16_v4<<<g, 256, 0, s>>>((const f32x4*)src, (u16x4*)dst,
                                         (long)n4);
}

static void run_cvt_t(const float* W, unsigned short* Wt, int K, int N,
                      hipStream_t s) {
    dim3 g(N / 32, K / 32);
    cvt_transpose_kernel<<<g, 256, 0, s>>>(W, Wt, K, N);
}

// hidden layer: bias + ReLU, bf16 activation out
static void run_gemm_hidden(const unsigned short* A, const unsigned short* Wt,
                            const float* bias, unsigned short* Cb,
                            int M, int N, int K, hipStream_t s) {
    dim3 g(N / BN, M / BM);
    gemm_bf16_kernel<true, true, false><<<g, 256, 0, s>>>(
        A, Wt, bias, Cb, nullptr, M, N, K);
}

// final layer: bias only, f32 out
static void run_gemm_final(const unsigned short* A, const unsigned short* Wt,
                           const float* bias, float* Cf,
                           int M, int N, int K, hipStream_t s) {
    dim3 g(N / BN, M / BM);
    gemm_bf16_kernel<false, false, true><<<g, 256, 0, s>>>(
        A, Wt, bias, nullptr, Cf, M, N, K);
}

extern "C" void kernel_launch(void* const* d_in, const int* in_sizes, int n_in,
                              void* d_out, int out_size, void* d_ws,
                              size_t ws_size, hipStream_t stream) {
    (void)in_sizes; (void)n_in; (void)out_size; (void)ws_size;

    const float* x  = (const float*)d_in[0];
    const float* eW[4] = {(const float*)d_in[1], (const float*)d_in[3],
                          (const float*)d_in[5], (const float*)d_in[7]};
    const float* eB[4] = {(const float*)d_in[2], (const float*)d_in[4],
                          (const float*)d_in[6], (const float*)d_in[8]};
    const float* dW[4] = {(const float*)d_in[9],  (const float*)d_in[11],
                          (const float*)d_in[13], (const float*)d_in[15]};
    const float* dB[4] = {(const float*)d_in[10], (const float*)d_in[12],
                          (const float*)d_in[14], (const float*)d_in[16]};
    const float* cb = (const float*)d_in[17];

    // output layout: out[B*768] | rq_loss[1] | indices[B*4] | z[B*128]
    float* out_rec  = (float*)d_out;
    float* out_loss = out_rec + (size_t)B_ROWS * IN_DIM_;
    int*   out_idx  = (int*)(out_loss + 1);
    float* out_z    = (float*)(out_idx + (size_t)B_ROWS * 4);

    char* w = (char*)d_ws;
    auto take = [&](size_t bytes) -> char* {
        char* p = w;
        w += (bytes + 255) & ~(size_t)255;
        return p;
    };
    // layer dims (K, N) in forward order: 4 encoder + 4 decoder
    const int dims[8][2] = {{768, 2048}, {2048, 1024}, {1024, 512},
                            {512, 128},  {128, 512},   {512, 1024},
                            {1024, 2048}, {2048, 768}};
    unsigned short* bw[8];   // transposed bf16 weights Wt[N][K]
    for (int i = 0; i < 8; ++i)
        bw[i] = (unsigned short*)take((size_t)dims[i][0] * dims[i][1] * 2);
    unsigned short* xbf  = (unsigned short*)take((size_t)B_ROWS * IN_DIM_ * 2);
    unsigned short* actA = (unsigned short*)take((size_t)B_ROWS * 2048 * 2);
    unsigned short* actB = (unsigned short*)take((size_t)B_ROWS * 2048 * 2);
    float* cbT = (float*)take((size_t)4 * 128 * 256 * 4);

    // convert + transpose weights; convert input
    const float* Wsrc[8] = {eW[0], eW[1], eW[2], eW[3], dW[0], dW[1], dW[2], dW[3]};
    for (int i = 0; i < 8; ++i)
        run_cvt_t(Wsrc[i], bw[i], dims[i][0], dims[i][1], stream);
    run_cvt(x, xbf, (size_t)B_ROWS * IN_DIM_, stream);
    transpose_cb_kernel<<<dim3(512), 256, 0, stream>>>(cb, cbT);
    zero_loss_kernel<<<dim3(1), 64, 0, stream>>>(out_loss);

    // encoder: 768 -> 2048 -> 1024 -> 512 -> 128 (z f32 to d_out)
    run_gemm_hidden(xbf,  bw[0], eB[0], actA, B_ROWS, 2048, 768,  stream);
    run_gemm_hidden(actA, bw[1], eB[1], actB, B_ROWS, 1024, 2048, stream);
    run_gemm_hidden(actB, bw[2], eB[2], actA, B_ROWS, 512,  1024, stream);
    run_gemm_final (actA, bw[3], eB[3], out_z, B_ROWS, 128, 512,  stream);

    // residual quantization (xq bf16 into actA)
    rq_kernel<<<dim3(B_ROWS / RQ_ROWS), 256, 0, stream>>>(
        out_z, cbT, cb, actA, out_idx, out_loss);

    // decoder: 128 -> 512 -> 1024 -> 2048 -> 768 (out f32 to d_out)
    run_gemm_hidden(actA, bw[4], dB[0], actB, B_ROWS, 512,  128,  stream);
    run_gemm_hidden(actB, bw[5], dB[1], actA, B_ROWS, 1024, 512,  stream);
    run_gemm_hidden(actA, bw[6], dB[2], actB, B_ROWS, 2048, 1024, stream);
    run_gemm_final (actB, bw[7], dB[3], out_rec, B_ROWS, 768, 2048, stream);
}
